// GSLRec_15401752724063
// MI455X (gfx1250) — matrix operand order repaired and checked by reading, not compiled, not run on hardware
//
#include <hip/hip_runtime.h>
#include <hip/hip_bf16.h>

typedef __attribute__((ext_vector_type(16))) _Float16 v16h;
typedef __attribute__((ext_vector_type(8)))  float    v8f;

#define DIMS 64

// ---------------- monotone float<->uint encode for atomicMax on floats ----------
__device__ __forceinline__ unsigned fenc(float f) {
    unsigned u = __float_as_uint(f);
    return (u & 0x80000000u) ? ~u : (u | 0x80000000u);
}
__device__ __forceinline__ float fdec(unsigned k) {
    unsigned u = (k & 0x80000000u) ? (k & 0x7FFFFFFFu) : ~k;
    return __uint_as_float(u);
}

// ---------------- concat user/item embeddings into h and acc --------------------
__global__ __launch_bounds__(256) void concat_kernel(
    const float* __restrict__ ue, const float* __restrict__ ie,
    float* __restrict__ h, float* __restrict__ acc, int usz, int total) {
    int i = blockIdx.x * blockDim.x + threadIdx.x;
    if (i >= total) return;
    float v = (i < usz) ? ue[i] : ie[i - usz];
    h[i] = v;
    acc[i] = v;
}

// ---------------- GCN edge scatter: hn[row] += val * h[col] ---------------------
__global__ __launch_bounds__(256) void gcn_edge_kernel(
    const float* __restrict__ h, const float* __restrict__ val,
    const int* __restrict__ row, const int* __restrict__ col,
    float* __restrict__ hn, int E) {
    int t = blockIdx.x * blockDim.x + threadIdx.x;
    if (t >= E * 16) return;
    int e = t >> 4;
    int q = (t & 15) << 2;
    float v = val[e];
    const float4 hv = *(const float4*)(h + (size_t)col[e] * DIMS + q);
    float* dst = hn + (size_t)row[e] * DIMS + q;
    atomicAdd(dst + 0, v * hv.x);
    atomicAdd(dst + 1, v * hv.y);
    atomicAdd(dst + 2, v * hv.z);
    atomicAdd(dst + 3, v * hv.w);
}

__global__ __launch_bounds__(256) void add_inplace_kernel(
    float* __restrict__ acc, const float* __restrict__ x, int n) {
    int i = blockIdx.x * blockDim.x + threadIdx.x;
    if (i < n) acc[i] += x[i];
}

// ---------------- WMMA GEMM: Y(n x dout) = X(n x 64) @ W(64 x dout) -------------
// One wave per 16x16 output tile; K=64 done as two f16 16x16x32 WMMAs.
__global__ __launch_bounds__(128) void wmma_gemm_kernel(
    const float* __restrict__ X, const float* __restrict__ W,
    float* __restrict__ Y, int n, int dout) {
    int ntiles = (n >> 4) * (dout >> 4);
    int tile = blockIdx.x * 4 + (threadIdx.x >> 5);
    if (tile >= ntiles) return;                 // uniform per wave: EXEC stays all-ones
    int tiles_n = dout >> 4;
    int m0 = (tile / tiles_n) << 4;
    int n0 = (tile % tiles_n) << 4;
    int lane = threadIdx.x & 31;
    int M = lane & 15;
    int hs = lane >> 4;

    v8f c = {};
#pragma unroll
    for (int kc = 0; kc < 2; ++kc) {
        int kbase = kc * 32;
        v16h a, b;
        const float* xrow = X + (size_t)(m0 + M) * DIMS + kbase;
#pragma unroll
        for (int j = 0; j < 16; ++j) {
            int vg = j >> 1;
            int k = ((vg < 4) ? (hs * 8 + 2 * vg) : (16 + hs * 8 + 2 * (vg - 4))) + (j & 1);
            a[j] = (_Float16)xrow[k];
        }
        const float* wrow = W + (size_t)kbase * dout + n0 + M;
#pragma unroll
        for (int j = 0; j < 16; ++j) b[j] = (_Float16)wrow[(size_t)((j < 8) ? (hs * 8 + j) : (8 + hs * 8 + j)) * dout];
        c = __builtin_amdgcn_wmma_f32_16x16x32_f16(false, a, false, b, (short)0, c, false, false);
    }
#pragma unroll
    for (int r = 0; r < 8; ++r)
        Y[(size_t)(m0 + r + 8 * hs) * dout + n0 + (lane & 15)] = c[r];
}

// ---------------- attention left/right scores -----------------------------------
__global__ __launch_bounds__(256) void elr_kernel(
    const float* __restrict__ h, const float* __restrict__ al, const float* __restrict__ ar,
    float* __restrict__ el, float* __restrict__ er, int n, int heads) {
    int t = blockIdx.x * blockDim.x + threadIdx.x;
    if (t >= n * heads) return;
    int hd = t % heads;
    const float* hp = h + (size_t)(t / heads) * heads * DIMS + hd * DIMS;
    float sl = 0.f, sr = 0.f;
#pragma unroll 8
    for (int d = 0; d < DIMS; ++d) {
        float v = hp[d];
        sl += v * al[hd * DIMS + d];
        sr += v * ar[hd * DIMS + d];
    }
    el[t] = sl;
    er[t] = sr;
}

// ---------------- edge score + segment max --------------------------------------
__global__ __launch_bounds__(256) void edge_max_kernel(
    const float* __restrict__ el, const float* __restrict__ er,
    const int* __restrict__ src, const int* __restrict__ dst,
    float* __restrict__ ebuf, unsigned* __restrict__ menc, int E, int heads) {
    int t = blockIdx.x * blockDim.x + threadIdx.x;
    if (t >= E * heads) return;
    int e = t / heads, hd = t % heads;
    float s = el[src[e] * heads + hd] + er[dst[e] * heads + hd];
    s = (s > 0.f) ? s : 0.2f * s;           // leaky_relu, slope 0.2
    ebuf[t] = s;
    atomicMax(&menc[dst[e] * heads + hd], fenc(s));
}

// ---------------- exp(e - m) + segment sum --------------------------------------
__global__ __launch_bounds__(256) void edge_exp_kernel(
    float* __restrict__ ebuf, const unsigned* __restrict__ menc,
    float* __restrict__ denom, const int* __restrict__ dst, int E, int heads) {
    int t = blockIdx.x * blockDim.x + threadIdx.x;
    if (t >= E * heads) return;
    int e = t / heads, hd = t % heads;
    float m = fdec(menc[dst[e] * heads + hd]);
    if (!__builtin_isfinite(m)) m = 0.f;
    float ex = expf(ebuf[t] - m);
    ebuf[t] = ex;
    atomicAdd(&denom[dst[e] * heads + hd], ex);
}

// ---------------- weighted aggregation out[dst] += alpha * h[src] ---------------
__global__ __launch_bounds__(256) void edge_agg_kernel(
    const float* __restrict__ h, const float* __restrict__ ebuf,
    const float* __restrict__ denom, const int* __restrict__ src,
    const int* __restrict__ dst, float* __restrict__ out, int E, int heads) {
    int t = blockIdx.x * blockDim.x + threadIdx.x;
    if (t >= E * heads * 16) return;
    int q = (t & 15) << 2;
    int eh = t >> 4;
    int e = eh / heads, hd = eh % heads;
    float alpha = ebuf[eh] / (denom[dst[e] * heads + hd] + 1e-16f);
    const float4 hv = *(const float4*)(h + (size_t)src[e] * heads * DIMS + hd * DIMS + q);
    float* op = out + (size_t)dst[e] * heads * DIMS + hd * DIMS + q;
    atomicAdd(op + 0, alpha * hv.x);
    atomicAdd(op + 1, alpha * hv.y);
    atomicAdd(op + 2, alpha * hv.z);
    atomicAdd(op + 3, alpha * hv.w);
}

// ---------------- elementwise elu (in place) ------------------------------------
__global__ __launch_bounds__(256) void elu_kernel(float* __restrict__ x, int n) {
    int i = blockIdx.x * blockDim.x + threadIdx.x;
    if (i >= n) return;
    float v = x[i];
    x[i] = (v > 0.f) ? v : expm1f(v);
}

// ---------------- mean over 4 heads ---------------------------------------------
__global__ __launch_bounds__(256) void mean_heads_kernel(
    const float* __restrict__ x, float* __restrict__ y, int n) {
    int i = blockIdx.x * blockDim.x + threadIdx.x;
    if (i >= n * DIMS) return;
    int node = i / DIMS, d = i % DIMS;
    const float* p = x + (size_t)node * 4 * DIMS + d;
    y[i] = 0.25f * (p[0] + p[DIMS] + p[2 * DIMS] + p[3 * DIMS]);
}

// ---------------- BPR scoring + reg loss ----------------------------------------
__global__ __launch_bounds__(256) void score_kernel(
    const float* __restrict__ acc, const float* __restrict__ users1,
    const float* __restrict__ items1, const float* __restrict__ user_emb,
    const float* __restrict__ item_emb, const int* __restrict__ user,
    const int* __restrict__ pos, const int* __restrict__ neg,
    float* __restrict__ out, int B, int U) {
    int b = blockIdx.x * blockDim.x + threadIdx.x;
    if (b >= B) return;
    int u = user[b], p = pos[b], g = neg[b];
    float ps = 0.f, ns = 0.f, reg = 0.f;
#pragma unroll 4
    for (int d = 0; d < DIMS; ++d) {
        float ue = acc[(size_t)u * DIMS + d] * 0.25f + users1[(size_t)u * DIMS + d];
        float pe = acc[(size_t)(U + p) * DIMS + d] * 0.25f + items1[(size_t)p * DIMS + d];
        float ne = acc[(size_t)(U + g) * DIMS + d] * 0.25f + items1[(size_t)g * DIMS + d];
        ps += ue * pe;
        ns += ue * ne;
        float a0 = user_emb[(size_t)u * DIMS + d];
        float a1 = item_emb[(size_t)p * DIMS + d];
        float a2 = item_emb[(size_t)g * DIMS + d];
        reg += a0 * a0 + a1 * a1 + a2 * a2;
    }
    float x = ns - ps;
    float sp = fmaxf(x, 0.f) + log1pf(expf(-fabsf(x)));  // softplus, overflow-safe
    atomicAdd(&out[0], sp / (float)B);
    atomicAdd(&out[1], 0.5f * reg / (float)B);
}

// ================================================================================
static inline int gridFor(long n, int b) { return (int)((n + b - 1) / b); }

static void run_gat(const float* x, const float* W0, const float* al0, const float* ar0,
                    const float* W1, const float* al1, const float* ar1,
                    const int* src, const int* dst, int E, int n,
                    float* hbuf, float* out1, float* out2,
                    float* el, float* er, unsigned* menc, float* denom, float* ebuf,
                    float* result, hipStream_t stream) {
    // ---- layer 1: heads = 1, dout = 64 ----
    {
        const int heads = 1, dout = DIMS;
        int ntiles = (n >> 4) * (dout >> 4);
        wmma_gemm_kernel<<<(ntiles + 3) / 4, 128, 0, stream>>>(x, W0, hbuf, n, dout);
        elr_kernel<<<gridFor((long)n * heads, 256), 256, 0, stream>>>(hbuf, al0, ar0, el, er, n, heads);
        hipMemsetAsync(menc, 0, (size_t)n * heads * 4, stream);
        hipMemsetAsync(denom, 0, (size_t)n * heads * 4, stream);
        hipMemsetAsync(out1, 0, (size_t)n * DIMS * 4, stream);
        edge_max_kernel<<<gridFor((long)E * heads, 256), 256, 0, stream>>>(el, er, src, dst, ebuf, menc, E, heads);
        edge_exp_kernel<<<gridFor((long)E * heads, 256), 256, 0, stream>>>(ebuf, menc, denom, dst, E, heads);
        edge_agg_kernel<<<gridFor((long)E * heads * 16, 256), 256, 0, stream>>>(hbuf, ebuf, denom, src, dst, out1, E, heads);
        elu_kernel<<<gridFor((long)n * DIMS, 256), 256, 0, stream>>>(out1, n * DIMS);
    }
    // ---- layer 2: heads = 4, dout_total = 256 ----
    {
        const int heads = 4, dout = 4 * DIMS;
        int ntiles = (n >> 4) * (dout >> 4);
        wmma_gemm_kernel<<<(ntiles + 3) / 4, 128, 0, stream>>>(out1, W1, hbuf, n, dout);
        elr_kernel<<<gridFor((long)n * heads, 256), 256, 0, stream>>>(hbuf, al1, ar1, el, er, n, heads);
        hipMemsetAsync(menc, 0, (size_t)n * heads * 4, stream);
        hipMemsetAsync(denom, 0, (size_t)n * heads * 4, stream);
        hipMemsetAsync(out2, 0, (size_t)n * (size_t)dout * 4, stream);
        edge_max_kernel<<<gridFor((long)E * heads, 256), 256, 0, stream>>>(el, er, src, dst, ebuf, menc, E, heads);
        edge_exp_kernel<<<gridFor((long)E * heads, 256), 256, 0, stream>>>(ebuf, menc, denom, dst, E, heads);
        edge_agg_kernel<<<gridFor((long)E * heads * 16, 256), 256, 0, stream>>>(hbuf, ebuf, denom, src, dst, out2, E, heads);
        mean_heads_kernel<<<gridFor((long)n * DIMS, 256), 256, 0, stream>>>(out2, result, n);
    }
}

extern "C" void kernel_launch(void* const* d_in, const int* in_sizes, int n_in,
                              void* d_out, int out_size, void* d_ws, size_t ws_size,
                              hipStream_t stream) {
    const float* user_emb = (const float*)d_in[0];
    const float* item_emb = (const float*)d_in[1];
    const float* adj_val  = (const float*)d_in[2];
    const float* u_W0  = (const float*)d_in[3];
    const float* u_al0 = (const float*)d_in[4];
    const float* u_ar0 = (const float*)d_in[5];
    const float* u_W1  = (const float*)d_in[6];
    const float* u_al1 = (const float*)d_in[7];
    const float* u_ar1 = (const float*)d_in[8];
    const float* i_W0  = (const float*)d_in[9];
    const float* i_al0 = (const float*)d_in[10];
    const float* i_ar0 = (const float*)d_in[11];
    const float* i_W1  = (const float*)d_in[12];
    const float* i_al1 = (const float*)d_in[13];
    const float* i_ar1 = (const float*)d_in[14];
    const int* adj_row = (const int*)d_in[15];
    const int* adj_col = (const int*)d_in[16];
    const int* uu_src  = (const int*)d_in[17];
    const int* uu_dst  = (const int*)d_in[18];
    const int* ii_src  = (const int*)d_in[19];
    const int* ii_dst  = (const int*)d_in[20];
    const int* user = (const int*)d_in[21];
    const int* pos  = (const int*)d_in[22];
    const int* neg  = (const int*)d_in[23];

    const int U = in_sizes[0] / DIMS;            // 100000
    const int I = in_sizes[1] / DIMS;            // 50000
    const int N = U + I;                         // 150000
    const int E_UI = in_sizes[2];                // 2000000
    const int E_UU = in_sizes[17];               // 800000
    const int E_II = in_sizes[19];               // 400000
    const int B = in_sizes[21];                  // 8192

    float* out = (float*)d_out;

    // ---- workspace layout (256B-aligned slabs) ----
    size_t off = 0;
    auto alloc = [&](size_t nfloats) -> float* {
        float* p = (float*)((char*)d_ws + off);
        off += ((nfloats * sizeof(float) + 255) / 256) * 256;
        return p;
    };
    float* acc  = alloc((size_t)N * DIMS);
    float* hA   = alloc((size_t)N * DIMS);
    float* hB   = alloc((size_t)N * DIMS);
    float* hbuf = alloc((size_t)U * 4 * DIMS);   // GEMM output, max U x 256
    float* out1 = alloc((size_t)U * DIMS);       // layer-1 output / layer-2 input
    float* out2 = alloc((size_t)U * 4 * DIMS);   // layer-2 output
    float* el   = alloc((size_t)U * 4);
    float* er   = alloc((size_t)U * 4);
    unsigned* menc = (unsigned*)alloc((size_t)U * 4);
    float* denom   = alloc((size_t)U * 4);
    float* ebuf    = alloc((size_t)E_UU * 4);    // per-edge scores, max E_UU x 4 heads
    float* users1  = alloc((size_t)U * DIMS);
    float* items1  = alloc((size_t)I * DIMS);
    (void)ws_size;

    hipMemsetAsync(out, 0, 2 * sizeof(float), stream);

    // ---- LightGCN propagation: 3 layers, acc = sum of layers ----
    concat_kernel<<<gridFor((long)N * DIMS, 256), 256, 0, stream>>>(
        user_emb, item_emb, hA, acc, U * DIMS, N * DIMS);
    float* hc = hA;
    float* hn = hB;
    for (int l = 0; l < 3; ++l) {
        hipMemsetAsync(hn, 0, (size_t)N * DIMS * sizeof(float), stream);
        gcn_edge_kernel<<<gridFor((long)E_UI * 16, 256), 256, 0, stream>>>(
            hc, adj_val, adj_row, adj_col, hn, E_UI);
        add_inplace_kernel<<<gridFor((long)N * DIMS, 256), 256, 0, stream>>>(acc, hn, N * DIMS);
        float* t = hc; hc = hn; hn = t;
    }

    // ---- user GAT and item GAT (share scratch, run sequentially) ----
    run_gat(user_emb, u_W0, u_al0, u_ar0, u_W1, u_al1, u_ar1,
            uu_src, uu_dst, E_UU, U,
            hbuf, out1, out2, el, er, menc, denom, ebuf, users1, stream);
    run_gat(item_emb, i_W0, i_al0, i_ar0, i_W1, i_al1, i_ar1,
            ii_src, ii_dst, E_II, I,
            hbuf, out1, out2, el, er, menc, denom, ebuf, items1, stream);

    // ---- BPR loss + reg loss ----
    score_kernel<<<gridFor(B, 256), 256, 0, stream>>>(
        acc, users1, items1, user_emb, item_emb, user, pos, neg, out, B, U);
}
